// NetE_73821897884214
// MI455X (gfx1250) — matrix operand order repaired and checked by reading, not compiled, not run on hardware
//
#include <hip/hip_runtime.h>

// ---------------------------------------------------------------------------
// MI455X (gfx1250, wave32).  Conv/deconv as implicit GEMM on
// v_wmma_f32_16x16x32_f16.  Activations live in a halo-padded f16 layout
// [B][C][H+2][W+2] (zeroed border) so every im2col tap is a valid address:
//  - conv (k=4 s=2 p=1): K = Cin*16, A-gather = aligned b32/b64 loads.
//  - deconv (k=4 s=2 p=1): sub-pixel decomposition into 4 parity classes
//    (blockIdx.z), each a stride-1 2x2 conv -> K = Cin*4 (no wasted taps).
// Register blocking MI x NJ (2x4): per K-chunk load 2 A + 4 B fragments for
// 8 WMMAs; B fragments reused across both M sub-tiles (halves L2 weight
// traffic).  Per-chunk addressing: fixed vector offsets + scalar base bump.
// BN train-mode batch stats via LDS tree reduction.  WS use ~155.5 MB.
// ---------------------------------------------------------------------------

typedef __attribute__((ext_vector_type(16))) _Float16 v16h;
typedef __attribute__((ext_vector_type(8)))  float    v8f;

#define EPSBN 1e-5f

// ---------------- zero an f16 buffer (as u32 words) ----------------
__global__ __launch_bounds__(256)
void zero_buf_k(unsigned* __restrict__ p, long long nWords) {
  long long i = (long long)blockIdx.x * 256 + threadIdx.x;
  const long long stride = (long long)gridDim.x * 256;
  for (; i < nWords; i += stride) p[i] = 0u;
}

// ---------------- x (B,3,64,64) f32 -> f16 padded (B,4,66,66) interior ----------------
__global__ __launch_bounds__(256)
void x_to_f16_pad_k(const float* __restrict__ x, _Float16* __restrict__ y,
                    long long total) {
  long long i = (long long)blockIdx.x * 256 + threadIdx.x;
  const long long stride = (long long)gridDim.x * 256;
  for (; i < total; i += stride) {            // i over B*4*64*64
    const int c = (int)((i >> 12) & 3);
    if (c >= 3) continue;                     // pad channel stays zero
    const int ww = (int)(i & 63);
    const int hh = (int)((i >> 6) & 63);
    const int b  = (int)(i >> 14);
    const float v = x[((long long)b * 3 + c) * 4096 + ((long long)hh << 6) + ww];
    y[(((long long)b * 4 + c) * 66 + hh + 1) * 66 + ww + 1] = (_Float16)v;
  }
}

// ---------------- conv weight swizzle (K = Cin*16) ----------------
// B-fragment: lane = N col (lane&15) + K-half group (lane>>4); half h ->
// K = kc*32 + 16*(h>=8) + 8*(lane>=16) + (h&7).  conv weights (Cout,Cin,4,4).
__global__ __launch_bounds__(256)
void swizzle_conv_k(const float* __restrict__ w, _Float16* __restrict__ o,
                    int Cout, int Cin, int nK, long long total) {
  const long long i = (long long)blockIdx.x * 256 + threadIdx.x;
  if (i >= total) return;
  const int h    = (int)(i & 15);
  const int lane = (int)((i >> 4) & 31);
  const long long rest = i >> 9;
  const int kc  = (int)(rest % nK);
  const int coT = (int)(rest / nK);
  const int n = coT * 16 + (lane & 15);
  const int k = kc * 32 + ((h >> 3) << 4) + ((lane >> 4) << 3) + (h & 7);
  const int ci = k >> 4, tap = k & 15, kh = tap >> 2, kw = tap & 3;
  float v = 0.f;
  if (ci < Cin && n < Cout) v = w[((n * Cin + ci) * 4 + kh) * 4 + kw];
  o[i] = (_Float16)v;
}

// ---------------- deconv weight swizzle: 4 parity classes, K = Cin*4 ----------------
// class (qh,qw): local k = ci*4 + th*2 + tw ; th=0 -> kh=qh+2, th=1 -> kh=qh
// (same for tw/kw).  torch deconv weights (Cin,Cout,4,4).
__global__ __launch_bounds__(256)
void swizzle_deconv_k(const float* __restrict__ w, _Float16* __restrict__ o,
                      int Cout, int Cin, int nK, long long perClass) {
  const long long i = (long long)blockIdx.x * 256 + threadIdx.x;
  if (i >= perClass * 4) return;
  const int cls = (int)(i / perClass);
  const long long r = i % perClass;
  const int qh = cls >> 1, qw = cls & 1;
  const int h    = (int)(r & 15);
  const int lane = (int)((r >> 4) & 31);
  const long long rest = r >> 9;
  const int kc  = (int)(rest % nK);
  const int coT = (int)(rest / nK);
  const int n = coT * 16 + (lane & 15);
  const int k = kc * 32 + ((h >> 3) << 4) + ((lane >> 4) << 3) + (h & 7);
  const int ci = k >> 2, th = (k >> 1) & 1, tw = k & 1;
  const int kh = qh + 2 - 2 * th, kw = qw + 2 - 2 * tw;
  float v = 0.f;
  if (ci < Cin && n < Cout) v = w[((ci * Cout + n) * 4 + kh) * 4 + kw];
  o[i] = (_Float16)v;
}

// ---------------- conv implicit GEMM (MI x NJ register blocking) ----------------
// One wave: MI*16 output positions x NJ*16 output channels.  A fragment rows:
// lanes 0-15 hold K {0..7,16..23}, lanes 16-31 hold K {8..15,24..31};
// per lane this is taps kh=2*hi,2*hi+1 (kw 0..3) for channels ci0,ci0+1.
template <int MI, int NJ>
__global__ __launch_bounds__(32)
void conv_gemm_k(const _Float16* __restrict__ in, const _Float16* __restrict__ wswz,
                 float* __restrict__ out,
                 int Cin, int Hp, int Wp, int logCout, int logHo, int logWo, int nK) {
  const int lane = threadIdx.x;
  const int hi = lane >> 4;
  const int lo = lane & 15;
  const int HpWp = Hp * Wp;
  const int Homask = (1 << logHo) - 1;
  const int Womask = (1 << logWo) - 1;

  // per-M-subtile gather offsets (padded coords always in-bounds)
  unsigned offA[MI], offB[MI], offC[MI], offD[MI];
#pragma unroll
  for (int mi = 0; mi < MI; ++mi) {
    const int m  = (blockIdx.x * MI + mi) * 16 + lo;
    const int ow = m & Womask;
    const int oh = (m >> logWo) & Homask;
    const int b  = m >> (logWo + logHo);
    offA[mi] = ((unsigned)(b * Cin * HpWp + (2 * oh + 2 * hi) * Wp + 2 * ow)) * 2u;
    offB[mi] = offA[mi] + (unsigned)Wp * 2u;       // kh row +1
    offC[mi] = offA[mi] + (unsigned)HpWp * 2u;     // channel ci0+1
    offD[mi] = offB[mi] + (unsigned)HpWp * 2u;
  }
  const unsigned step = (unsigned)(4 * HpWp);      // bytes per K-chunk (2 ch)

  v8f acc[MI * NJ] = {};
  const long long wbase = (long long)blockIdx.y * NJ;
  const char* inb = (const char*)in;

  for (int kc = 0; kc < nK; ++kc) {
    __builtin_prefetch(wswz + ((wbase * nK + kc + 1) * 32 + lane) * 16, 0, 3);
    union { v16h v; unsigned u[8]; } A[MI];
#pragma unroll
    for (int mi = 0; mi < MI; ++mi) {
      A[mi].u[0] = *(const unsigned*)(inb + offA[mi]);
      A[mi].u[1] = *(const unsigned*)(inb + offA[mi] + 4);
      A[mi].u[2] = *(const unsigned*)(inb + offB[mi]);
      A[mi].u[3] = *(const unsigned*)(inb + offB[mi] + 4);
      A[mi].u[4] = *(const unsigned*)(inb + offC[mi]);
      A[mi].u[5] = *(const unsigned*)(inb + offC[mi] + 4);
      A[mi].u[6] = *(const unsigned*)(inb + offD[mi]);
      A[mi].u[7] = *(const unsigned*)(inb + offD[mi] + 4);
    }
    inb += step;
#pragma unroll
    for (int j = 0; j < NJ; ++j) {
      const v16h bf = *(const v16h*)(wswz + (((wbase + j) * nK + kc) * 32 + lane) * 16);
#pragma unroll
      for (int mi = 0; mi < MI; ++mi)
        acc[mi * NJ + j] = __builtin_amdgcn_wmma_f32_16x16x32_f16(
            false, A[mi].v, false, bf, (short)0, acc[mi * NJ + j], false, false);
    }
  }

  // store D (lane: N = lo; VGPR r: M = r + 8*hi)
#pragma unroll
  for (int mi = 0; mi < MI; ++mi) {
#pragma unroll
    for (int r = 0; r < 8; ++r) {
      const int M   = (blockIdx.x * MI + mi) * 16 + r + (hi << 3);
      const int sow = M & Womask;
      const int soh = (M >> logWo) & Homask;
      const int sb  = M >> (logWo + logHo);
#pragma unroll
      for (int j = 0; j < NJ; ++j) {
        const int co = blockIdx.y * (NJ * 16) + j * 16 + lo;
        const long long oidx =
            ((((((long long)sb << logCout) + co) << logHo) + soh) << logWo) + sow;
        out[oidx] = acc[mi * NJ + j][r];
      }
    }
  }
}

// ---------------- deconv implicit GEMM (sub-pixel classes, MI x NJ) ----------------
// blockIdx.z = parity class (qh,qw).  Positions (b,ohh,oww) over Hin x Win;
// oh = 2*ohh + (1-qh), ow = 2*oww + (1-qw).  Each output reads a 2x2 input
// patch whose padded top-left is (ohh+1-qh, oww+1-qw).  K = Cin*4.
template <int MI, int NJ, bool FINAL>
__global__ __launch_bounds__(32)
void deconv_gemm_k(const _Float16* __restrict__ in, const _Float16* __restrict__ wswz,
                   float* __restrict__ out,
                   int Cin, int Hp, int Wp, int Cout, int logCout,
                   int logOhh, int logOww, int nK, long long perClassW) {
  const int lane = threadIdx.x;
  const int hi = lane >> 4;
  const int lo = lane & 15;
  const int HpWp = Hp * Wp;
  const int hmask = (1 << logOhh) - 1;
  const int wmask = (1 << logOww) - 1;
  const int qh = (int)blockIdx.z >> 1, qw = (int)blockIdx.z & 1;

  // fixed per-half byte offsets per M-subtile; per-chunk advance is scalar
  unsigned aoff[MI][16];
#pragma unroll
  for (int mi = 0; mi < MI; ++mi) {
    const int m   = (blockIdx.x * MI + mi) * 16 + lo;
    const int oww = m & wmask;
    const int ohh = (m >> logOww) & hmask;
    const int b   = m >> (logOww + logOhh);
    const int rowTop  = ohh + 1 - qh;   // padded row of tap th=0
    const int colLeft = oww + 1 - qw;   // padded col of tap tw=0
#pragma unroll
    for (int h = 0; h < 16; ++h) {
      const int kl = ((h >> 3) << 4) + (h & 7) + (hi << 3);
      const int ci = kl >> 2;           // ci_local 0..7
      const int th = (kl >> 1) & 1, tw = kl & 1;
      aoff[mi][h] =
          ((unsigned)((b * Cin + ci) * HpWp + (rowTop + th) * Wp + colLeft + tw)) * 2u;
    }
  }
  const unsigned step = (unsigned)(16 * HpWp);  // bytes per K-chunk (8 ch)

  v8f acc[MI * NJ] = {};
  const _Float16* wcls = wswz + (long long)blockIdx.z * perClassW;
  const long long wbase = (long long)blockIdx.y * NJ;
  const char* inb = (const char*)in;

  for (int kc = 0; kc < nK; ++kc) {
    __builtin_prefetch(wcls + ((wbase * nK + kc + 1) * 32 + lane) * 16, 0, 3);
    v16h a[MI];
#pragma unroll
    for (int mi = 0; mi < MI; ++mi)
#pragma unroll
      for (int h = 0; h < 16; ++h)
        a[mi][h] = *(const _Float16*)(inb + aoff[mi][h]);
    inb += step;
#pragma unroll
    for (int j = 0; j < NJ; ++j) {
      const v16h bf = *(const v16h*)(wcls + (((wbase + j) * nK + kc) * 32 + lane) * 16);
#pragma unroll
      for (int mi = 0; mi < MI; ++mi)
        acc[mi * NJ + j] = __builtin_amdgcn_wmma_f32_16x16x32_f16(
            false, a[mi], false, bf, (short)0, acc[mi * NJ + j], false, false);
    }
  }

#pragma unroll
  for (int mi = 0; mi < MI; ++mi) {
#pragma unroll
    for (int r = 0; r < 8; ++r) {
      const int M    = (blockIdx.x * MI + mi) * 16 + r + (hi << 3);
      const int sww  = M & wmask;
      const int shh  = (M >> logOww) & hmask;
      const int sb   = M >> (logOww + logOhh);
      const int soh  = 2 * shh + 1 - qh;
      const int sow  = 2 * sww + 1 - qw;
#pragma unroll
      for (int j = 0; j < NJ; ++j) {
        const int co = blockIdx.y * (NJ * 16) + j * 16 + lo;
        if (FINAL) {
          if (co < Cout) {  // Cout=3, Ho=Wo=64
            const long long oidx = (((long long)sb * 3 + co) << 12) + (soh << 6) + sow;
            out[oidx] = tanhf(acc[mi * NJ + j][r]);
          }
        } else {
          const long long oidx =
              ((((((long long)sb << logCout) + co) << (logOhh + 1)) + soh)
               << (logOww + 1)) + sow;
          out[oidx] = acc[mi * NJ + j][r];
        }
      }
    }
  }
}

// ---------------- BN training-mode stats (one block per channel) ----------------
__global__ __launch_bounds__(256)
void bn_stats_k(const float* __restrict__ x, float* __restrict__ mean,
                float* __restrict__ var, int C, int HW, int B) {
  const int c = blockIdx.x;
  const int tid = threadIdx.x;
  float s = 0.f, s2 = 0.f;
  for (int bb = 0; bb < B; ++bb) {
    const float* xp = x + ((long long)bb * C + c) * HW;
    for (int p = tid; p < HW; p += 256) {
      const float v = xp[p];
      s += v; s2 += v * v;
    }
  }
  __shared__ float sh1[256], sh2[256];
  sh1[tid] = s; sh2[tid] = s2;
  __syncthreads();
  for (int o = 128; o > 0; o >>= 1) {
    if (tid < o) { sh1[tid] += sh1[tid + o]; sh2[tid] += sh2[tid + o]; }
    __syncthreads();
  }
  if (tid == 0) {
    const float inv = 1.f / (float)((long long)B * HW);
    const float mm = sh1[0] * inv;
    mean[c] = mm;
    var[c]  = sh2[0] * inv - mm * mm;  // biased variance
  }
}

// ---------------- BN normalize + activation -> halo-padded f16 ----------------
__global__ __launch_bounds__(256)
void bn_act_pad_k(const float* __restrict__ x, const float* __restrict__ mean,
                  const float* __restrict__ var, const float* __restrict__ g,
                  const float* __restrict__ bb, _Float16* __restrict__ y,
                  int logH, int logW, int logC, long long total, int relu_mode) {
  const int W = 1 << logW, H = 1 << logH;
  const int Wp = W + 2, Hp = H + 2;
  long long i = (long long)blockIdx.x * 256 + threadIdx.x;
  const long long stride = (long long)gridDim.x * 256;
  for (; i < total; i += stride) {
    const int ww = (int)(i & (W - 1));
    const int hh = (int)((i >> logW) & (H - 1));
    const int c  = (int)((i >> (logW + logH)) & ((1 << logC) - 1));
    const int b  = (int)(i >> (logW + logH + logC));
    float v = (x[i] - mean[c]) * rsqrtf(var[c] + EPSBN) * g[c] + bb[c];
    v = relu_mode ? fmaxf(v, 0.f) : (v >= 0.f ? v : 0.2f * v);
    y[((((long long)b << logC) + c) * Hp + hh + 1) * Wp + ww + 1] = (_Float16)v;
  }
}

// ---------------- channel-wise Linear(16,16) on padded 6x6 layout ----------------
__global__ __launch_bounds__(256)
void chanlin_k(const _Float16* __restrict__ h, const float* __restrict__ w,
               const float* __restrict__ bias, _Float16* __restrict__ y) {
  const int i = blockIdx.x * 256 + threadIdx.x;  // exactly 256*512*16 threads
  const int t = i & 15;
  const int c = (i >> 4) & 511;
  const int b = i >> 13;
  float acc = bias[c * 16 + t];
  const _Float16* hp = h + ((long long)b * 512 + c) * 36;
  const float* wp = w + ((long long)c * 16 + t) * 16;
#pragma unroll
  for (int s = 0; s < 16; ++s)
    acc += (float)hp[(1 + (s >> 2)) * 6 + 1 + (s & 3)] * wp[s];
  y[((long long)b * 512 + c) * 36 + (1 + (t >> 2)) * 6 + 1 + (t & 3)] = (_Float16)acc;
}

// ---------------------------------------------------------------------------
extern "C" void kernel_launch(void* const* d_in, const int* in_sizes, int n_in,
                              void* d_out, int out_size, void* d_ws, size_t ws_size,
                              hipStream_t stream) {
  (void)in_sizes; (void)n_in; (void)out_size; (void)ws_size;

  const float* x    = (const float*)d_in[0];
  const float* w1   = (const float*)d_in[1];
  const float* w2   = (const float*)d_in[2];
  const float* w3   = (const float*)d_in[3];
  const float* w4   = (const float*)d_in[4];
  const float* g1   = (const float*)d_in[5];
  const float* b1   = (const float*)d_in[6];
  const float* g2   = (const float*)d_in[7];
  const float* b2   = (const float*)d_in[8];
  const float* g3   = (const float*)d_in[9];
  const float* b3   = (const float*)d_in[10];
  const float* g4   = (const float*)d_in[11];
  const float* b4   = (const float*)d_in[12];
  const float* cw_w = (const float*)d_in[13];
  const float* cw_b = (const float*)d_in[14];
  const float* dw1  = (const float*)d_in[15];
  const float* dw2  = (const float*)d_in[16];
  const float* dw3  = (const float*)d_in[17];
  const float* dw4  = (const float*)d_in[18];
  const float* dg1  = (const float*)d_in[19];
  const float* db1  = (const float*)d_in[20];
  const float* dg2  = (const float*)d_in[21];
  const float* db2  = (const float*)d_in[22];
  const float* dg3  = (const float*)d_in[23];
  const float* db3  = (const float*)d_in[24];

  // ---- workspace layout (~155.5 MB) ----
  char* ws = (char*)d_ws;
  float*    F32S = (float*)(ws);                        // 67,108,864 B
  _Float16* F16A = (_Float16*)(ws + 67108864);          // 37,879,808 B (padded act)
  _Float16* F16B = (_Float16*)(ws + 104988672);         // 37,879,808 B
  _Float16* WQ   = (_Float16*)(ws + 142868480);         // 12,582,912 B reserved
  float*    MEAN = (float*)(ws + 155451392);            // 512 floats
  float*    VAR  = MEAN + 512;
  const long long F16WORDS = 37879808 / 4;

  // swizzled-weight element offsets
  const long long O_C1 = 0,       O_C2 = 4096,    O_C3 = 135168,  O_C4 = 659456;
  const long long O_D1 = 2756608, O_D2 = 4853760, O_D3 = 5378048, O_D4 = 5509120;
  const long long S_D1 = 524288,  S_D2 = 131072,  S_D3 = 32768,   S_D4 = 4096;

  const int B = 256;

  auto zero = [&](_Float16* p) {
    zero_buf_k<<<8192, 256, 0, stream>>>((unsigned*)p, F16WORDS);
  };
  auto bnact = [&](const float* cx, const float* g, const float* bb, _Float16* y,
                   int logC, int logH, int relu) {
    const int C = 1 << logC, HW = 1 << (2 * logH);
    bn_stats_k<<<C, 256, 0, stream>>>(cx, MEAN, VAR, C, HW, B);
    zero(y);
    const long long total = (long long)B * C * HW;
    bn_act_pad_k<<<(unsigned)((total + 255) / 256), 256, 0, stream>>>(
        cx, MEAN, VAR, g, bb, y, logH, logH, logC, total, relu);
  };

  // ---- prep: padded x, swizzled weights ----
  zero(F16A);
  x_to_f16_pad_k<<<16384, 256, 0, stream>>>(x, F16A, (long long)B * 4 * 4096);
  auto swzc = [&](const float* w, long long off, int Cout, int Cin, int nK) {
    const long long tot = (long long)(Cout / 16) * nK * 512;
    swizzle_conv_k<<<(unsigned)((tot + 255) / 256), 256, 0, stream>>>(
        w, WQ + off, Cout, Cin, nK, tot);
  };
  auto swzd = [&](const float* w, long long off, int Cout, int Cin, int nK,
                  long long perClass) {
    swizzle_deconv_k<<<(unsigned)((perClass * 4 + 255) / 256), 256, 0, stream>>>(
        w, WQ + off, Cout, Cin, nK, perClass);
  };
  swzc(w1,  O_C1,  64,   3,   2);
  swzc(w2,  O_C2, 128,  64,  32);
  swzc(w3,  O_C3, 256, 128,  64);
  swzc(w4,  O_C4, 512, 256, 128);
  swzd(dw1, O_D1, 256, 512, 64, S_D1);
  swzd(dw2, O_D2, 128, 256, 32, S_D2);
  swzd(dw3, O_D3,  64, 128, 16, S_D3);
  swzd(dw4, O_D4,   3,  64,  8, S_D4);   // Cout padded 3 -> 16

  // ---- encoder (MI=2, NJ=4: 32 positions x 64 channels per wave) ----
  conv_gemm_k<2, 4><<<dim3(B * 32 * 32 / 32, 1), 32, 0, stream>>>(
      F16A, WQ + O_C1, F32S, 4, 66, 66, 6, 5, 5, 2);
  bnact(F32S, g1, b1, F16B, 6, 5, 0);

  conv_gemm_k<2, 4><<<dim3(B * 16 * 16 / 32, 2), 32, 0, stream>>>(
      F16B, WQ + O_C2, F32S, 64, 34, 34, 7, 4, 4, 32);
  bnact(F32S, g2, b2, F16A, 7, 4, 0);

  conv_gemm_k<2, 4><<<dim3(B * 8 * 8 / 32, 4), 32, 0, stream>>>(
      F16A, WQ + O_C3, F32S, 128, 18, 18, 8, 3, 3, 64);
  bnact(F32S, g3, b3, F16B, 8, 3, 0);

  conv_gemm_k<2, 4><<<dim3(B * 4 * 4 / 32, 8), 32, 0, stream>>>(
      F16B, WQ + O_C4, F32S, 256, 10, 10, 9, 2, 2, 128);
  bnact(F32S, g4, b4, F16A, 9, 2, 0);

  // ---- channel-wise linear (padded 6x6 in, padded 6x6 out) ----
  zero(F16B);
  chanlin_k<<<(B * 512 * 16) / 256, 256, 0, stream>>>(F16A, cw_w, cw_b, F16B);

  // ---- decoder (4 parity classes per layer via blockIdx.z) ----
  deconv_gemm_k<2, 4, false><<<dim3(B * 4 * 4 / 32, 4, 4), 32, 0, stream>>>(
      F16B, WQ + O_D1, F32S, 512, 6, 6, 256, 8, 2, 2, 64, S_D1);
  bnact(F32S, dg1, db1, F16A, 8, 3, 1);

  deconv_gemm_k<2, 4, false><<<dim3(B * 8 * 8 / 32, 2, 4), 32, 0, stream>>>(
      F16A, WQ + O_D2, F32S, 256, 10, 10, 128, 7, 3, 3, 32, S_D2);
  bnact(F32S, dg2, db2, F16B, 7, 4, 1);

  deconv_gemm_k<2, 4, false><<<dim3(B * 16 * 16 / 32, 1, 4), 32, 0, stream>>>(
      F16B, WQ + O_D3, F32S, 128, 18, 18, 64, 6, 4, 4, 16, S_D3);
  bnact(F32S, dg3, db3, F16A, 6, 5, 1);

  // final: Cout padded 3->16 in weights, store co<3 with tanh -> d_out f32 NCHW
  deconv_gemm_k<2, 1, true><<<dim3(B * 32 * 32 / 32, 1, 4), 32, 0, stream>>>(
      F16A, WQ + O_D4, (float*)d_out, 64, 34, 34, 3, 0, 5, 5, 8, S_D4);
}
